// TextClassifier_33217277067654
// MI455X (gfx1250) — matrix operand order repaired and checked by reading, not compiled, not run on hardware
//
#include <hip/hip_runtime.h>
#include <hip/hip_bf16.h>
#include <math.h>

typedef _Float16 half_t;
typedef __attribute__((ext_vector_type(16))) _Float16 v16h;
typedef __attribute__((ext_vector_type(8)))  _Float16 v8h;
typedef __attribute__((ext_vector_type(8)))  float    v8f;

#define WMMA(a, b, c) __builtin_amdgcn_wmma_f32_16x16x32_f16(false, (a), false, (b), (short)0, (c), false, false)

static constexpr int Bsz = 8, Seq = 2048, Win = 64;
static constexpr int Tok = Bsz * Seq;   // 16384

// ---- A-operand loader: 16x32 f16 tile, row-major with row stride ldk (halves)
// Contiguous k order, as in the B loaders: lane l<16 holds row l, K = 0..15; lane l+16 holds row l, K = 16..31.
__device__ __forceinline__ v16h load_A16(const half_t* __restrict__ tile, int ldk, int lane) {
    int r  = lane & 15;
    int kb = (lane >> 4) * 16;
    const v8h* p0 = (const v8h*)(tile + (size_t)r * ldk + kb);
    const v8h* p1 = (const v8h*)(tile + (size_t)r * ldk + 8 + kb);
    v8h lo = *p0, hi = *p1;
    v16h a;
#pragma unroll
    for (int i = 0; i < 8; i++) { a[i] = lo[i]; a[i + 8] = hi[i]; }
    return a;
}

// ---- B-operand loader: 32x16 f16 tile where B[k][n] is contiguous in k.
// rowptr = &data[this lane's column][k0]; lane holds column n=lane&15, K=(lane>>4)*16 + 0..15.
__device__ __forceinline__ v16h load_Brow(const half_t* __restrict__ rowptr, int lane) {
    int kb = (lane >> 4) * 16;
    const v8h* p = (const v8h*)(rowptr + kb);
    v8h lo = p[0], hi = p[1];
    v16h b;
#pragma unroll
    for (int i = 0; i < 8; i++) { b[i] = lo[i]; b[i + 8] = hi[i]; }
    return b;
}

// ---- B-operand loader where the k-offset (kb) is already folded into the pointer
__device__ __forceinline__ v16h load_Bcontig(const half_t* __restrict__ p16) {
    const v8h* p = (const v8h*)p16;
    v8h lo = p[0], hi = p[1];
    v16h b;
#pragma unroll
    for (int i = 0; i < 8; i++) { b[i] = lo[i]; b[i + 8] = hi[i]; }
    return b;
}

// ---------------- f32 -> f16 convert ----------------
__global__ void cvt_kernel(const float* __restrict__ in, half_t* __restrict__ out, int n) {
    int i = blockIdx.x * blockDim.x + threadIdx.x;
    if (i < n) out[i] = (half_t)in[i];
}

// ---------------- embedding gather -> f16 ----------------
__global__ __launch_bounds__(256) void embed_kernel(const int* __restrict__ text,
                                                    const float* __restrict__ emb,
                                                    half_t* __restrict__ x16) {
    int tkn = blockIdx.x, e = threadIdx.x;
    int id = text[tkn];
    x16[(size_t)tkn * 256 + e] = (half_t)emb[(size_t)id * 256 + e];
}

// ---------------- WMMA GEMM: C[M,N] = A[M,K](f16) * W[N,K]^T(f16) + bias ----------------
// mode 0: f32 output
// mode 2: f16 output; cols<256 scaled by 1/sqrt(64) (q); cols>=512 (v) written TRANSPOSED
//         into vT[b*256 + (n-512)][s]  (i.e. [B*H*64][Seq]) for the attention PV stage.
__global__ __launch_bounds__(256) void gemm_wmma(const half_t* __restrict__ A,
                                                 const half_t* __restrict__ W,
                                                 const float* __restrict__ bias,
                                                 half_t* __restrict__ outH,
                                                 float* __restrict__ outF,
                                                 half_t* __restrict__ vT,
                                                 int M, int N, int K, int mode) {
    int lane = threadIdx.x & 31;
    int wave = blockIdx.x * (blockDim.x >> 5) + (threadIdx.x >> 5);
    int ntg = N >> 6;                       // each wave: 1 m-tile x 4 n-tiles (16x64 strip)
    int mt = wave / ntg, ng = wave - mt * ntg;
    if (mt >= (M >> 4)) return;             // grids divide exactly; wave-uniform guard
    int m0 = mt << 4, n0 = ng << 6;
    int nn = lane & 15, hi = lane >> 4;

    v8f c0 = {}, c1 = {}, c2 = {}, c3 = {};
    for (int k0 = 0; k0 < K; k0 += 32) {
        v16h a  = load_A16(A + (size_t)m0 * K + k0, K, lane);
        v16h b0 = load_Brow(W + (size_t)(n0 +  0 + nn) * K + k0, lane);
        v16h b1 = load_Brow(W + (size_t)(n0 + 16 + nn) * K + k0, lane);
        v16h b2 = load_Brow(W + (size_t)(n0 + 32 + nn) * K + k0, lane);
        v16h b3 = load_Brow(W + (size_t)(n0 + 48 + nn) * K + k0, lane);
        c0 = WMMA(a, b0, c0);
        c1 = WMMA(a, b1, c1);
        c2 = WMMA(a, b2, c2);
        c3 = WMMA(a, b3, c3);
    }
    v8f cc[4] = {c0, c1, c2, c3};
#pragma unroll
    for (int t = 0; t < 4; t++) {
        int n = n0 + t * 16 + nn;
        float bv = bias[n];
        float scale = (mode == 2 && n < 256) ? 0.125f : 1.0f;
#pragma unroll
        for (int g = 0; g < 8; g++) {
            int m = m0 + g + hi * 8;        // C layout: VGPR g -> row g (+8 for lanes 16..31)
            float v = (cc[t][g] + bv) * scale;
            if (mode == 0) {
                outF[(size_t)m * N + n] = v;
            } else if (mode == 2 && n >= 512) {
                // v-part: store transposed [b*256 + h*64 + d][s]
                int hd = n - 512;
                int bb = m >> 11, ss = m & (Seq - 1);
                vT[((size_t)bb * 256 + hd) * (size_t)Seq + ss] = (half_t)v;
            } else {
                outH[(size_t)m * N + n] = (half_t)v;
            }
        }
    }
}

// ---------------- banded flash attention (window 64), all-WMMA ----------------
// qkv: [token][768] f16, q at +0 (pre-scaled), k at +256 (v-region unused).
// vT : [b*256 + h*64 + d][s] f16 (transposed V for contiguous B-operand loads).
__global__ __launch_bounds__(128) void attn_wmma(const half_t* __restrict__ qkv,
                                                 const half_t* __restrict__ vT,
                                                 half_t* __restrict__ aout) {
    int lane = threadIdx.x & 31;
    int w = threadIdx.x >> 5;
    int tile = blockIdx.x * 4 + w;          // b*512 + h*128 + qt
    int qt = tile & 127;
    int h  = (tile >> 7) & 3;
    int b  = tile >> 9;
    int q0 = qt << 4;

    __shared__ __align__(16) half_t Pbuf[4][16 * 32];
    half_t* P = Pbuf[w];

    const size_t tb = (size_t)b * Seq;
    const half_t* Qb  = qkv + tb * 768 + (size_t)h * 64;
    const half_t* Kb  = Qb + 256;
    const half_t* VTb = vT + ((size_t)b * 256 + (size_t)h * 64) * Seq;
    int col = lane & 15, hi = lane >> 4;

    // Q tile as two A operands (head dim 64 = 2 x K32 steps)
    v16h qa0 = load_A16(Qb + (size_t)q0 * 768,      768, lane);
    v16h qa1 = load_A16(Qb + (size_t)q0 * 768 + 32, 768, lane);

    v8f o0 = {}, o1 = {}, o2 = {}, o3 = {};
    float mrow[8], lrow[8];
#pragma unroll
    for (int g = 0; g < 8; g++) { mrow[g] = -1e30f; lrow[g] = 0.f; }

    int kbeg = q0 - Win;      if (kbeg < 0)  kbeg = 0;
    int kend = q0 + 16 + Win; if (kend > Seq) kend = Seq;

    for (int c0 = kbeg; c0 < kend; c0 += 32) {
        // ---- scores S[16 x 32] = Q x K^T (two 16-col tiles, two K32 steps each)
        int key0 = c0 + col;
        int key1 = c0 + 16 + col;
        int k0c = key0 < Seq ? key0 : Seq - 1;
        int k1c = key1 < Seq ? key1 : Seq - 1;
        v16h kb00 = load_Brow(Kb + (size_t)k0c * 768,      lane);
        v16h kb01 = load_Brow(Kb + (size_t)k0c * 768 + 32, lane);
        v16h kb10 = load_Brow(Kb + (size_t)k1c * 768,      lane);
        v16h kb11 = load_Brow(Kb + (size_t)k1c * 768 + 32, lane);
        v8f s0 = {}, s1 = {};
        s0 = WMMA(qa0, kb00, s0); s0 = WMMA(qa1, kb01, s0);
        s1 = WMMA(qa0, kb10, s1); s1 = WMMA(qa1, kb11, s1);

        // ---- online softmax per row (rows live across the 16 lanes of each half-wave)
#pragma unroll
        for (int g = 0; g < 8; g++) {
            int qi = q0 + g + hi * 8;
            bool ok0 = (key0 < kend) && (key0 >= qi - Win) && (key0 <= qi + Win);
            bool ok1 = (key1 < kend) && (key1 >= qi - Win) && (key1 <= qi + Win);
            float v0 = ok0 ? s0[g] : -1e30f;
            float v1 = ok1 ? s1[g] : -1e30f;
            float mx = fmaxf(v0, v1);
#pragma unroll
            for (int d = 1; d < 16; d <<= 1) mx = fmaxf(mx, __shfl_xor(mx, d, 32));
            float mnew  = fmaxf(mrow[g], mx);
            float alpha = __expf(mrow[g] - mnew);
            float p0 = ok0 ? __expf(v0 - mnew) : 0.f;
            float p1 = ok1 ? __expf(v1 - mnew) : 0.f;
            float rs = p0 + p1;
#pragma unroll
            for (int d = 1; d < 16; d <<= 1) rs += __shfl_xor(rs, d, 32);
            lrow[g] = lrow[g] * alpha + rs;
            mrow[g] = mnew;
            o0[g] *= alpha; o1[g] *= alpha; o2[g] *= alpha; o3[g] *= alpha;
            int r = g + hi * 8;
            P[r * 32 + col]      = (half_t)p0;   // stage P in LDS to re-shape into A layout
            P[r * 32 + 16 + col] = (half_t)p1;
        }
        asm volatile("s_wait_dscnt 0" ::: "memory");   // CDNA5 split counter: LDS store->load
        v16h pa = load_A16(P, 32, lane);

        // ---- O += P x V  (vT rows are contiguous in key -> b128 loads)
        // 16-key blocks are 16-aligned and kend is 16-aligned, so a block is either fully
        // valid or fully masked (p==0); clamping only mis-maps fully-masked blocks.
        int kstart = c0 + hi * 16;
        if (kstart > Seq - 16) kstart = Seq - 16;
#pragma unroll
        for (int t = 0; t < 4; t++) {
            v16h bv = load_Bcontig(VTb + (size_t)(t * 16 + col) * Seq + kstart);
            if      (t == 0) o0 = WMMA(pa, bv, o0);
            else if (t == 1) o1 = WMMA(pa, bv, o1);
            else if (t == 2) o2 = WMMA(pa, bv, o2);
            else             o3 = WMMA(pa, bv, o3);
        }
    }

    // ---- normalize and store f16 [token][256]
#pragma unroll
    for (int g = 0; g < 8; g++) {
        float inv = 1.0f / lrow[g];
        int tok = q0 + g + hi * 8;
        size_t base = (tb + tok) * 256 + (size_t)h * 64;
        aout[base +  0 + col] = (half_t)(o0[g] * inv);
        aout[base + 16 + col] = (half_t)(o1[g] * inv);
        aout[base + 32 + col] = (half_t)(o2[g] * inv);
        aout[base + 48 + col] = (half_t)(o3[g] * inv);
    }
}

// ---------------- max-pool over sequence: two-stage tree ----------------
__global__ __launch_bounds__(256) void maxpool1(const float* __restrict__ o,
                                                float* __restrict__ part) {
    int b = blockIdx.x >> 4, c = blockIdx.x & 15, e = threadIdx.x;
    const float* p = o + ((size_t)b * Seq + (size_t)c * 128) * 256 + e;
    float m = -3.4e38f;
    for (int s = 0; s < 128; s++) m = fmaxf(m, p[(size_t)s * 256]);
    part[((size_t)b * 16 + c) * 256 + e] = m;
}
__global__ __launch_bounds__(256) void maxpool2(const float* __restrict__ part,
                                                float* __restrict__ ctx) {
    int b = blockIdx.x, e = threadIdx.x;
    float m = -3.4e38f;
#pragma unroll
    for (int c = 0; c < 16; c++) m = fmaxf(m, part[((size_t)b * 16 + c) * 256 + e]);
    ctx[b * 256 + e] = m;
}

// ---------------- tiny MLP head (8 rows; scalar f32) ----------------
__global__ __launch_bounds__(256) void head_kernel(const float* __restrict__ ctx,
        const float* __restrict__ f1w, const float* __restrict__ f1b,
        const float* __restrict__ f2w, const float* __restrict__ f2b,
        const float* __restrict__ f3w, const float* __restrict__ f3b,
        float* __restrict__ out) {
    __shared__ float sc[256], sh1[512], sh2[256];
    int b = blockIdx.x, t = threadIdx.x;
    sc[t] = ctx[b * 256 + t];
    __syncthreads();
    for (int j = t; j < 512; j += 256) {
        const float* w = f1w + (size_t)j * 256;
        float acc = f1b[j];
        for (int k = 0; k < 256; k++) acc += sc[k] * w[k];
        sh1[j] = acc > 0.f ? acc : 0.01f * acc;
    }
    __syncthreads();
    {
        const float* w = f2w + (size_t)t * 512;
        float acc = f2b[t];
        for (int k = 0; k < 512; k++) acc += sh1[k] * w[k];
        sh2[t] = acc > 0.f ? acc : 0.01f * acc;
    }
    __syncthreads();
    if (t < 20) {
        const float* w = f3w + (size_t)t * 256;
        float acc = f3b[t];
        for (int k = 0; k < 256; k++) acc += sh2[k] * w[k];
        out[b * 20 + t] = acc;
    }
}

extern "C" void kernel_launch(void* const* d_in, const int* in_sizes, int n_in,
                              void* d_out, int out_size, void* d_ws, size_t ws_size,
                              hipStream_t stream) {
    const int*   text = (const int*)d_in[0];
    const float* emb  = (const float*)d_in[1];
    const float* inw  = (const float*)d_in[2];
    const float* inb  = (const float*)d_in[3];
    const float* ow   = (const float*)d_in[4];
    const float* ob   = (const float*)d_in[5];
    const float* f1w  = (const float*)d_in[6];
    const float* f1b  = (const float*)d_in[7];
    const float* f2w  = (const float*)d_in[8];
    const float* f2b  = (const float*)d_in[9];
    const float* f3w  = (const float*)d_in[10];
    const float* f3b  = (const float*)d_in[11];
    float* out = (float*)d_out;
    (void)in_sizes; (void)n_in; (void)out_size; (void)ws_size;

    char* ws = (char*)d_ws;
    size_t off = 0;
    auto alloc = [&](size_t bytes) -> char* {
        char* p = ws + off;
        off = (off + bytes + 255) & ~(size_t)255;
        return p;
    };
    half_t* x16   = (half_t*)alloc((size_t)Tok * 256 * 2);        //  8 MB
    half_t* wq16  = (half_t*)alloc((size_t)768 * 256 * 2);        //  0.4 MB
    half_t* wo16  = (half_t*)alloc((size_t)256 * 256 * 2);        //  0.13 MB
    half_t* qkv16 = (half_t*)alloc((size_t)Tok * 768 * 2);        // 24 MB (v region unused)
    half_t* vT    = (half_t*)alloc((size_t)Bsz * 256 * Seq * 2);  //  8 MB  [B*H*64][Seq]
    half_t* at16  = (half_t*)alloc((size_t)Tok * 256 * 2);        //  8 MB
    float*  oproj = (float*)alloc((size_t)Tok * 256 * 4);         // 16 MB
    float*  part  = (float*)alloc((size_t)Bsz * 16 * 256 * 4);    //  0.13 MB
    float*  ctx   = (float*)alloc((size_t)Bsz * 256 * 4);

    // 1) weight f32->f16
    cvt_kernel<<<(768 * 256 + 255) / 256, 256, 0, stream>>>(inw, wq16, 768 * 256);
    cvt_kernel<<<(256 * 256 + 255) / 256, 256, 0, stream>>>(ow,  wo16, 256 * 256);
    // 2) embedding gather -> f16
    embed_kernel<<<Tok, 256, 0, stream>>>(text, emb, x16);
    // 3) fused QKV projection (q pre-scaled, v transposed); 1024*12 waves / 8 per block
    gemm_wmma<<<1536, 256, 0, stream>>>(x16, wq16, inb, qkv16, nullptr, vT, Tok, 768, 256, 2);
    // 4) banded flash attention: 8*4*128 = 4096 q-tiles, 4 waves/block
    attn_wmma<<<1024, 128, 0, stream>>>(qkv16, vT, at16);
    // 5) output projection -> f32; 1024*4 waves / 8 per block
    gemm_wmma<<<512, 256, 0, stream>>>(at16, wo16, ob, nullptr, oproj, nullptr, Tok, 256, 256, 0);
    // 6) max-pool over sequence (two-stage)
    maxpool1<<<Bsz * 16, 256, 0, stream>>>(oproj, part);
    maxpool2<<<Bsz, 256, 0, stream>>>(part, ctx);
    // 7) MLP head
    head_kernel<<<Bsz, 256, 0, stream>>>(ctx, f1w, f1b, f2w, f2b, f3w, f3b, out);
}
